// FP_Linear_6030134083893
// MI455X (gfx1250) — compile-verified, matrix operand order checked by reading, not run on hardware
//
#include <hip/hip_runtime.h>
#include <hip/hip_bf16.h>

// ---------------------------------------------------------------------------
// Types for WMMA fragments (gfx1250, wave32)
// ---------------------------------------------------------------------------
typedef __attribute__((ext_vector_type(16))) _Float16 v16h;
typedef __attribute__((ext_vector_type(8)))  _Float16 v8h;
typedef __attribute__((ext_vector_type(8)))  float    v8f;

union Frag16 { v16h v; v8h h[2]; };

#define T_DIM 16384   // rows of x (GEMM M)
#define K_DIM 1024    // reduction dim
#define N_DIM 1024    // output cols (weight rows)

// ---------------------------------------------------------------------------
// Ordered-uint encoding: monotonic map float -> uint so uint atomicMin/Max
// implement float min/max.
// ---------------------------------------------------------------------------
__device__ __forceinline__ unsigned int f2ord(float f) {
    unsigned int u = __float_as_uint(f);
    return (u & 0x80000000u) ? ~u : (u | 0x80000000u);
}
__device__ __forceinline__ float ord2f(unsigned int u) {
    return (u & 0x80000000u) ? __uint_as_float(u & 0x7fffffffu)
                             : __uint_as_float(~u);
}

__global__ void init_minmax_kernel(unsigned int* mm) {
    if (threadIdx.x == 0) {
        mm[0] = 0u;           // running max (ordered space minimum)
        mm[1] = 0xffffffffu;  // running min (ordered space maximum)
    }
}

// Grid-stride min/max reduction over the weight matrix.
__global__ void minmax_kernel(const float* __restrict__ w, int n,
                              unsigned int* __restrict__ mm) {
    __shared__ float smax[256];
    __shared__ float smin[256];
    float vmax = -3.402823466e+38f;
    float vmin =  3.402823466e+38f;
    for (int i = blockIdx.x * blockDim.x + threadIdx.x; i < n;
         i += gridDim.x * blockDim.x) {
        float v = w[i];
        vmax = fmaxf(vmax, v);
        vmin = fminf(vmin, v);
    }
    smax[threadIdx.x] = vmax;
    smin[threadIdx.x] = vmin;
    __syncthreads();
    for (int s = 128; s > 0; s >>= 1) {
        if (threadIdx.x < s) {
            smax[threadIdx.x] = fmaxf(smax[threadIdx.x], smax[threadIdx.x + s]);
            smin[threadIdx.x] = fminf(smin[threadIdx.x], smin[threadIdx.x + s]);
        }
        __syncthreads();
    }
    if (threadIdx.x == 0) {
        atomicMax(&mm[0], f2ord(smax[0]));
        atomicMin(&mm[1], f2ord(smin[0]));
    }
}

// 4-bit asymmetric fake-quant of W, result cast to f16.
__global__ void quant_weight_kernel(const float* __restrict__ w,
                                    const unsigned int* __restrict__ mm,
                                    _Float16* __restrict__ wh, int n) {
    float wmax = ord2f(mm[0]);
    float wmin = ord2f(mm[1]);
    float alpha = wmax - wmin;
    float beta  = wmin;
    const float step = 15.0f;  // 2^4 - 1
    int i = blockIdx.x * blockDim.x + threadIdx.x;
    if (i < n) {
        float v = w[i];
        float ws = (v - beta) / alpha;
        float R  = roundf(step * ws) / step;
        float wq = alpha * R + beta;
        wq = (v != 0.0f) ? wq : 0.0f;  // zeros stay zero
        wh[i] = (_Float16)wq;
    }
}

// fp32 -> f16 cast of x (one pass; afterwards everything is L2 resident).
__global__ void cvt_x_kernel(const float* __restrict__ x,
                             _Float16* __restrict__ xh, int n) {
    for (int i = blockIdx.x * blockDim.x + threadIdx.x; i < n;
         i += gridDim.x * blockDim.x) {
        xh[i] = (_Float16)x[i];
    }
}

// ---------------------------------------------------------------------------
// WMMA GEMM: out[t][n] = sum_k xh[t][k] * wh[n][k] + bias[n]
// Block = 256 threads = 8 waves arranged 2(M) x 4(N).
// Wave tile = 32(M) x 64(N): 2 A-frags, 4 B-frags, 8 v8f accumulators.
// Block tile = 64(M) x 256(N). Grid = (N/256, T/64).
//
// 16-bit A-frag layout (ISA 7.12.2): lane&15 = row M, lane>>4 picks K-halves:
//   halves 0..7  <- K = base+0..7,  halves 8..15 <- K = base+16..23,
//   where base = (lane>>4)*8. Two contiguous 16B loads per lane.
// B (K x N) sourced from wh row n (contiguous in K) with the same striping.
// ---------------------------------------------------------------------------
__global__ void __launch_bounds__(256, 2)
gemm_wmma_kernel(const _Float16* __restrict__ xh,
                 const _Float16* __restrict__ wh,
                 const float* __restrict__ bias,
                 float* __restrict__ out) {
    const int lane   = threadIdx.x & 31;
    const int waveId = threadIdx.x >> 5;
    const int l  = lane & 15;
    const int hi = lane >> 4;

    const int wm = waveId >> 2;          // 0..1
    const int wn = waveId & 3;           // 0..3

    const int bt  = blockIdx.y * 64;     // block M base
    const int bn  = blockIdx.x * 256;    // block N base
    const int wt  = bt + wm * 32;        // wave M base
    const int wnb = bn + wn * 64;        // wave N base

    v8f acc[2][4];
#pragma unroll
    for (int i = 0; i < 2; ++i)
#pragma unroll
        for (int j = 0; j < 4; ++j)
            acc[i][j] = (v8f)0.0f;

    // Per-lane base pointers (K striping via hi)
    const _Float16* arow[2];
#pragma unroll
    for (int i = 0; i < 2; ++i)
        arow[i] = xh + (size_t)(wt + i * 16 + l) * K_DIM + hi * 8;
    const _Float16* brow[4];
#pragma unroll
    for (int j = 0; j < 4; ++j)
        brow[j] = wh + (size_t)(wnb + j * 16 + l) * K_DIM + hi * 8;

#pragma unroll 4
    for (int k0 = 0; k0 < K_DIM; k0 += 32) {
        Frag16 a[2];
#pragma unroll
        for (int i = 0; i < 2; ++i) {
            a[i].h[0] = *(const v8h*)(arow[i] + k0);
            a[i].h[1] = *(const v8h*)(arow[i] + k0 + 16);
        }
        Frag16 b[4];
#pragma unroll
        for (int j = 0; j < 4; ++j) {
            b[j].h[0] = *(const v8h*)(brow[j] + k0);
            b[j].h[1] = *(const v8h*)(brow[j] + k0 + 16);
        }
#pragma unroll
        for (int i = 0; i < 2; ++i)
#pragma unroll
            for (int j = 0; j < 4; ++j)
                acc[i][j] = __builtin_amdgcn_wmma_f32_16x16x32_f16(
                    /*neg_a=*/false, a[i].v,
                    /*neg_b=*/false, b[j].v,
                    /*c_mod=*/(short)0, acc[i][j],
                    /*reuse_a=*/false, /*reuse_b=*/false);
    }

    // Epilogue: C/D layout -> lane = N (lane&15), VGPR r = row r (lanes 0-15)
    // or row 8+r (lanes 16-31). Add bias and store fp32.
    float bv[4];
#pragma unroll
    for (int j = 0; j < 4; ++j) bv[j] = bias[wnb + j * 16 + l];

#pragma unroll
    for (int i = 0; i < 2; ++i) {
#pragma unroll
        for (int j = 0; j < 4; ++j) {
            const int col = wnb + j * 16 + l;
#pragma unroll
            for (int r = 0; r < 8; ++r) {
                const int row = wt + i * 16 + hi * 8 + r;
                out[(size_t)row * N_DIM + col] = acc[i][j][r] + bv[j];
            }
        }
    }
}

// ---------------------------------------------------------------------------
extern "C" void kernel_launch(void* const* d_in, const int* in_sizes, int n_in,
                              void* d_out, int out_size, void* d_ws, size_t ws_size,
                              hipStream_t stream) {
    const float* x    = (const float*)d_in[0];  // [16384, 1024]
    const float* w    = (const float*)d_in[1];  // [1024, 1024]
    const float* bias = (const float*)d_in[2];  // [1024]
    float* out = (float*)d_out;                 // [16384, 1024]

    char* ws = (char*)d_ws;
    unsigned int* mm = (unsigned int*)ws;                         // 2 uints
    _Float16* wh = (_Float16*)(ws + 256);                         // 2 MiB
    _Float16* xh = (_Float16*)(ws + 256 + (size_t)N_DIM * K_DIM * 2); // 32 MiB

    const int wn = N_DIM * K_DIM;   // 1,048,576
    const int xn = T_DIM * K_DIM;   // 16,777,216

    init_minmax_kernel<<<1, 32, 0, stream>>>(mm);
    minmax_kernel<<<256, 256, 0, stream>>>(w, wn, mm);
    quant_weight_kernel<<<(wn + 255) / 256, 256, 0, stream>>>(w, mm, wh, wn);
    cvt_x_kernel<<<8192, 256, 0, stream>>>(x, xh, xn);

    dim3 grid(N_DIM / 256, T_DIM / 64);   // (4, 256)
    gemm_wmma_kernel<<<grid, 256, 0, stream>>>(xh, wh, bias, out);
}
